// RelativeMultiHeadAttention_45586782879947
// MI455X (gfx1250) — matrix operand order repaired and checked by reading, not compiled, not run on hardware
//
#include <hip/hip_runtime.h>
#include <hip/hip_bf16.h>
#include <stdint.h>

// ---------------------------------------------------------------------------
// Relative multi-head attention for MI455X (gfx1250), all matmuls via
// v_wmma_f32_16x16x32_bf16 (wave32). Flash-style attention: scores never
// materialized; rel-shift handled as a per-wave 16x96 positional band GEMM.
// K/V tiles and bf16 A tiles staged with GLOBAL_LOAD_ASYNC_TO_LDS_B128
// (ASYNCcnt) -- CDNA5 async global->LDS path, no VGPR round trip.
// ---------------------------------------------------------------------------

#define EDIM   1024
#define HEADS  16
#define HDIM   64
#define BSZ    4
#define LSEQ   1024
#define NTOK   (BSZ * LSEQ)          // 4096
#define NREL   (2 * LSEQ - 1)        // 2047

typedef __attribute__((ext_vector_type(16))) __bf16 v16bf;
typedef __attribute__((ext_vector_type(8)))  float  v8f;

__device__ __forceinline__ v8f wmma_bf16(v16bf a, v16bf b, v8f c) {
  // D = A(16x32) * B(32x16) + C, fp32 accumulate
  return __builtin_amdgcn_wmma_f32_16x16x32_bf16(false, a, false, b,
                                                 (short)0, c, false, false);
}

// ---- CDNA5 async global->LDS copy (16 B per lane, tracked by ASYNCcnt) ----
__device__ __forceinline__ void async_ld_b128(unsigned int lds_off,
                                              unsigned long long gaddr) {
  asm volatile("global_load_async_to_lds_b128 %0, %1, off"
               :: "v"(lds_off), "v"(gaddr) : "memory");
}
__device__ __forceinline__ void wait_async0() {
  asm volatile("s_wait_asynccnt 0" ::: "memory");
}

// ---------------- fragment loaders (contiguous k order, same for A and B) -------------
// A 16x32 bf16: lanes 0-15 -> M=lane, K=0..15 contiguous;
//               lanes 16-31 -> K=16..31.
__device__ __forceinline__ v16bf fragA_bf16(const __bf16* p, int row0, int ld, int k0) {
  const int lane = threadIdx.x & 31;
  const int m  = row0 + (lane & 15);
  const int kb = (lane >> 4) * 16;
  const __bf16* r = p + (size_t)m * ld + k0;
  v16bf a;
#pragma unroll
  for (int j = 0; j < 8; ++j) a[j] = r[kb + j];
#pragma unroll
  for (int j = 0; j < 8; ++j) a[8 + j] = r[8 + kb + j];
  return a;
}

__device__ __forceinline__ v16bf fragA_f32(const float* p, int row0, int ld, int k0, int rmax) {
  const int lane = threadIdx.x & 31;
  int m = row0 + (lane & 15);
  if (m > rmax) m = rmax;
  const int kb = (lane >> 4) * 16;
  const float* r = p + (size_t)m * ld + k0;
  v16bf a;
#pragma unroll
  for (int j = 0; j < 8; ++j) a[j] = (__bf16)r[kb + j];
#pragma unroll
  for (int j = 0; j < 8; ++j) a[8 + j] = (__bf16)r[8 + kb + j];
  return a;
}

// B 32x16 bf16, source stored "n-major" (row-major [N][K], i.e. B transposed
// in memory: weight rows W[n][k], K-tile rows Kt[key][d], Rk rows).
// lanes 0-15 -> N=lane, K=0..15 contiguous; lanes 16-31 -> K=16..31.
__device__ __forceinline__ v16bf fragB_nmajor_bf16(const __bf16* p, int n0, int ld, int k0) {
  const int lane = threadIdx.x & 31;
  const int n  = n0 + (lane & 15);
  const int kb = k0 + (lane >> 4) * 16;
  const __bf16* r = p + (size_t)n * ld + kb;
  v16bf b;
#pragma unroll
  for (int j = 0; j < 16; ++j) b[j] = r[j];
  return b;
}

__device__ __forceinline__ v16bf fragB_nmajor_bf16_clamp(const __bf16* p, int n0, int ld,
                                                         int k0, int nmax) {
  const int lane = threadIdx.x & 31;
  int n = n0 + (lane & 15);
  if (n < 0) n = 0;
  if (n > nmax) n = nmax;
  const int kb = k0 + (lane >> 4) * 16;
  const __bf16* r = p + (size_t)n * ld + kb;
  v16bf b;
#pragma unroll
  for (int j = 0; j < 16; ++j) b[j] = r[j];
  return b;
}

__device__ __forceinline__ v16bf fragB_nmajor_f32(const float* p, int n0, int ld,
                                                  int k0, int nmax) {
  const int lane = threadIdx.x & 31;
  int n = n0 + (lane & 15);
  if (n > nmax) n = nmax;
  const int kb = k0 + (lane >> 4) * 16;
  const float* r = p + (size_t)n * ld + kb;
  v16bf b;
#pragma unroll
  for (int j = 0; j < 16; ++j) b[j] = (__bf16)r[j];
  return b;
}

// B 32x16 bf16, source stored "k-major" (row-major [K][N], e.g. V tile [key][d]).
__device__ __forceinline__ v16bf fragB_kmajor_bf16(const __bf16* p, int k0, int ld, int n0) {
  const int lane = threadIdx.x & 31;
  const int n  = n0 + (lane & 15);
  const int kb = k0 + (lane >> 4) * 16;
  v16bf b;
#pragma unroll
  for (int j = 0; j < 16; ++j) b[j] = p[(size_t)(kb + j) * ld + n];
  return b;
}

// reductions across the 16 lanes of a half-wave (C/D rows live in half-waves)
__device__ __forceinline__ float redmax16(float v) {
  v = fmaxf(v, __shfl_xor(v, 1, 32));
  v = fmaxf(v, __shfl_xor(v, 2, 32));
  v = fmaxf(v, __shfl_xor(v, 4, 32));
  v = fmaxf(v, __shfl_xor(v, 8, 32));
  return v;
}
__device__ __forceinline__ float redsum16(float v) {
  v += __shfl_xor(v, 1, 32);
  v += __shfl_xor(v, 2, 32);
  v += __shfl_xor(v, 4, 32);
  v += __shfl_xor(v, 8, 32);
  return v;
}

union BF4 { __bf16 h[4]; unsigned long long u; };

// ---------------------------------------------------------------------------
// GEMM 1: out[n, e] = sum_k X[n,k] * W[e,k] + bias[e]
// X fp32 [NTOK, EDIM], W fp32 [EDIM, EDIM]; out bf16 scattered to [B,H,L,D].
// Block: 256 threads (8 waves), tile 128 rows x 64 cols, K chunks of 32.
// (fp32->bf16 conversion must go through VALU, so staging stays VGPR-based.)
// ---------------------------------------------------------------------------
__global__ __launch_bounds__(256) void gemm_f32in_bf16out(
    const float* __restrict__ X, const float* __restrict__ W,
    const float* __restrict__ bias, __bf16* __restrict__ out) {
  __shared__ __align__(16) __bf16 At[128 * 32];
  __shared__ __align__(16) __bf16 Bt[64 * 32];
  const int tid  = threadIdx.x;
  const int wave = tid >> 5;
  const int lane = tid & 31;
  const int hi   = lane >> 4;
  const int row0 = blockIdx.x * 128;
  const int e0   = blockIdx.y * 64;

  const v8f vz = {0.f, 0.f, 0.f, 0.f, 0.f, 0.f, 0.f, 0.f};
  v8f acc[4] = {vz, vz, vz, vz};

  for (int kc = 0; kc < EDIM; kc += 32) {
    __syncthreads();
    // stage A: 128x32 fp32 -> bf16 (1024 float4, 4 per thread, coalesced)
#pragma unroll
    for (int it = 0; it < 4; ++it) {
      const int v  = tid + it * 256;
      const int r  = v >> 3, c4 = v & 7;
      const float4 f = reinterpret_cast<const float4*>(X + (size_t)(row0 + r) * EDIM + kc)[c4];
      BF4 t4;
      t4.h[0] = (__bf16)f.x; t4.h[1] = (__bf16)f.y;
      t4.h[2] = (__bf16)f.z; t4.h[3] = (__bf16)f.w;
      *reinterpret_cast<unsigned long long*>(&At[r * 32 + c4 * 4]) = t4.u;
    }
    // stage B: 64 rows of W (e0..e0+63) x 32 cols (512 float4, 2 per thread)
#pragma unroll
    for (int it = 0; it < 2; ++it) {
      const int v  = tid + it * 256;
      const int r  = v >> 3, c4 = v & 7;
      const float4 f = reinterpret_cast<const float4*>(W + (size_t)(e0 + r) * EDIM + kc)[c4];
      BF4 t4;
      t4.h[0] = (__bf16)f.x; t4.h[1] = (__bf16)f.y;
      t4.h[2] = (__bf16)f.z; t4.h[3] = (__bf16)f.w;
      *reinterpret_cast<unsigned long long*>(&Bt[r * 32 + c4 * 4]) = t4.u;
    }
    __syncthreads();
    const v16bf a = fragA_bf16(At, wave * 16, 32, 0);
#pragma unroll
    for (int t = 0; t < 4; ++t) {
      const v16bf b = fragB_nmajor_bf16(Bt, t * 16, 32, 0);
      acc[t] = wmma_bf16(a, b, acc[t]);
    }
  }

  // epilogue: bias + scatter to [B, H, L, D]
#pragma unroll
  for (int t = 0; t < 4; ++t) {
    const int eg = e0 + t * 16 + (lane & 15);
    const float bv = bias[eg];
    const int h = eg >> 6, d = eg & 63;
#pragma unroll
    for (int r = 0; r < 8; ++r) {
      const int m  = r + 8 * hi;
      const int ng = row0 + wave * 16 + m;
      const int bb = ng >> 10, l = ng & (LSEQ - 1);
      out[(((size_t)bb * HEADS + h) * LSEQ + l) * HDIM + d] = (__bf16)(acc[t][r] + bv);
    }
  }
}

// ---------------------------------------------------------------------------
// GEMM 2: out fp32 [NTOK, EDIM] = ctx(bf16) @ Wo^T + bo  (final projection)
// A tile is a raw bf16 copy -> staged with async global->LDS (ASYNCcnt).
// ---------------------------------------------------------------------------
__global__ __launch_bounds__(256) void gemm_bf16in_f32out(
    const __bf16* __restrict__ Xbf, const float* __restrict__ W,
    const float* __restrict__ bias, float* __restrict__ out) {
  __shared__ __align__(16) __bf16 At[128 * 32];
  __shared__ __align__(16) __bf16 Bt[64 * 32];
  const int tid  = threadIdx.x;
  const int wave = tid >> 5;
  const int lane = tid & 31;
  const int hi   = lane >> 4;
  const int row0 = blockIdx.x * 128;
  const int e0   = blockIdx.y * 64;

  const unsigned int atBase = (unsigned int)(size_t)(void*)At;  // LDS offset

  const v8f vz = {0.f, 0.f, 0.f, 0.f, 0.f, 0.f, 0.f, 0.f};
  v8f acc[4] = {vz, vz, vz, vz};

  for (int kc = 0; kc < EDIM; kc += 32) {
    __syncthreads();
    // stage A: 128x32 bf16 async copy (512 x 16B, 2 per thread)
#pragma unroll
    for (int it = 0; it < 2; ++it) {
      const int v = tid + it * 256;
      const int r = v >> 2, c = v & 3;
      const unsigned long long g =
          (unsigned long long)(size_t)(Xbf + (size_t)(row0 + r) * EDIM + kc) + (unsigned)c * 16u;
      async_ld_b128(atBase + (unsigned)v * 16u, g);
    }
    // stage B: W rows fp32 -> bf16 (needs VALU convert)
#pragma unroll
    for (int it = 0; it < 2; ++it) {
      const int v  = tid + it * 256;
      const int r  = v >> 3, c4 = v & 7;
      const float4 f = reinterpret_cast<const float4*>(W + (size_t)(e0 + r) * EDIM + kc)[c4];
      BF4 t4;
      t4.h[0] = (__bf16)f.x; t4.h[1] = (__bf16)f.y;
      t4.h[2] = (__bf16)f.z; t4.h[3] = (__bf16)f.w;
      *reinterpret_cast<unsigned long long*>(&Bt[r * 32 + c4 * 4]) = t4.u;
    }
    wait_async0();
    __syncthreads();
    const v16bf a = fragA_bf16(At, wave * 16, 32, 0);
#pragma unroll
    for (int t = 0; t < 4; ++t) {
      const v16bf b = fragB_nmajor_bf16(Bt, t * 16, 32, 0);
      acc[t] = wmma_bf16(a, b, acc[t]);
    }
  }

#pragma unroll
  for (int t = 0; t < 4; ++t) {
    const int eg = e0 + t * 16 + (lane & 15);
    const float bv = bias[eg];
#pragma unroll
    for (int r = 0; r < 8; ++r) {
      const int m  = r + 8 * hi;
      const int ng = row0 + wave * 16 + m;
      out[(size_t)ng * EDIM + eg] = acc[t][r] + bv;
    }
  }
}

// ---------------------------------------------------------------------------
// Rk = rel_emb @ Wp^T  -> bf16 [NREL, HDIM].  One wave per 16-row tile.
// ---------------------------------------------------------------------------
__global__ __launch_bounds__(32) void relproj_kernel(
    const float* __restrict__ R, const float* __restrict__ Wp,
    __bf16* __restrict__ Rk) {
  const int row0 = blockIdx.x * 16;
  const int lane = threadIdx.x & 31;
  const int hi   = lane >> 4;

  const v8f vz = {0.f, 0.f, 0.f, 0.f, 0.f, 0.f, 0.f, 0.f};
  v8f acc[4] = {vz, vz, vz, vz};

#pragma unroll
  for (int ks = 0; ks < 2; ++ks) {
    const v16bf a = fragA_f32(R, row0, HDIM, 32 * ks, NREL - 1);
#pragma unroll
    for (int t = 0; t < 4; ++t) {
      const v16bf b = fragB_nmajor_f32(Wp, t * 16, HDIM, 32 * ks, HDIM - 1);
      acc[t] = wmma_bf16(a, b, acc[t]);
    }
  }
#pragma unroll
  for (int t = 0; t < 4; ++t) {
    const int o = t * 16 + (lane & 15);
#pragma unroll
    for (int r = 0; r < 8; ++r) {
      const int row = row0 + r + 8 * hi;
      if (row < NREL) Rk[(size_t)row * HDIM + o] = (__bf16)acc[t][r];
    }
  }
}

// ---------------------------------------------------------------------------
// Flash attention with relative-position band.
// Grid: (16 q-tiles, 64 b*h).  Block: 128 threads = 4 waves; wave w owns
// query rows q0w..q0w+15.  K/V tiles (64 keys, 8 KB contiguous each) staged
// with async global->LDS; positional band 16x96 per wave (rel idx j-i+15).
// ---------------------------------------------------------------------------
__global__ __launch_bounds__(128) void attn_kernel(
    const __bf16* __restrict__ Qbf, const __bf16* __restrict__ Kbf,
    const __bf16* __restrict__ Vbf, const __bf16* __restrict__ Rk,
    const unsigned char* __restrict__ maskp, __bf16* __restrict__ ctx) {
  __shared__ __align__(16) __bf16 Kt[64 * HDIM];      // 8 KB
  __shared__ __align__(16) __bf16 Vt[64 * HDIM];      // 8 KB
  __shared__ __align__(16) float  band[4][16][96];    // 24 KB
  __shared__ __align__(16) __bf16 Pst[4][16 * 64];    // 8 KB

  const int tid  = threadIdx.x;
  const int wave = tid >> 5;
  const int lane = tid & 31;
  const int hi   = lane >> 4;
  const int qt   = blockIdx.x;        // 0..15
  const int bh   = blockIdx.y;        // 0..63
  const int b    = bh >> 4;
  const int h    = bh & 15;
  const int q0w  = qt * 64 + wave * 16;
  const float scale = 0.125f;         // 1/sqrt(64)

  const __bf16* Qhead = Qbf + (size_t)bh * LSEQ * HDIM;
  const __bf16* Khead = Kbf + (size_t)bh * LSEQ * HDIM;
  const __bf16* Vhead = Vbf + (size_t)bh * LSEQ * HDIM;

  const unsigned int ktBase = (unsigned int)(size_t)(void*)Kt;
  const unsigned int vtBase = (unsigned int)(size_t)(void*)Vt;

  // Q fragments are loop-invariant per wave
  const v16bf a0 = fragA_bf16(Qhead, q0w, HDIM, 0);
  const v16bf a1 = fragA_bf16(Qhead, q0w, HDIM, 32);

  const v8f vz = {0.f, 0.f, 0.f, 0.f, 0.f, 0.f, 0.f, 0.f};
  v8f O[4] = {vz, vz, vz, vz};
  float mrun[8], lrun[8];
#pragma unroll
  for (int r = 0; r < 8; ++r) { mrun[r] = -1e30f; lrun[r] = 0.f; }

  for (int kt = 0; kt < LSEQ / 64; ++kt) {
    const int k0g = kt * 64;
    __syncthreads();
    // stage K and V tiles via async global->LDS (8 KB contiguous each,
    // 16 B per lane, 4 rounds x 128 threads)
    {
      const unsigned long long kG =
          (unsigned long long)(size_t)(Khead + (size_t)k0g * HDIM) + (unsigned)tid * 16u;
      const unsigned long long vG =
          (unsigned long long)(size_t)(Vhead + (size_t)k0g * HDIM) + (unsigned)tid * 16u;
#pragma unroll
      for (int it = 0; it < 4; ++it) {
        async_ld_b128(ktBase + (unsigned)tid * 16u + (unsigned)it * 2048u, kG + (unsigned)it * 2048u);
        async_ld_b128(vtBase + (unsigned)tid * 16u + (unsigned)it * 2048u, vG + (unsigned)it * 2048u);
      }
    }
    wait_async0();
    __syncthreads();

    // ---- content scores: S(16x64) = Q * K^T ----
    v8f S[4];
#pragma unroll
    for (int t = 0; t < 4; ++t) {
      S[t] = wmma_bf16(a0, fragB_nmajor_bf16(Kt, t * 16, HDIM, 0), vz);
      S[t] = wmma_bf16(a1, fragB_nmajor_bf16(Kt, t * 16, HDIM, 32), S[t]);
    }

    // ---- positional band: 16x96, r0w chosen so rel idx = jloc - m + 15 ----
    const int r0w = k0g - q0w - 15 + (LSEQ - 1);
#pragma unroll
    for (int rt = 0; rt < 6; ++rt) {
      v8f Pb = wmma_bf16(a0, fragB_nmajor_bf16_clamp(Rk, r0w + rt * 16, HDIM, 0, NREL - 1), vz);
      Pb = wmma_bf16(a1, fragB_nmajor_bf16_clamp(Rk, r0w + rt * 16, HDIM, 32, NREL - 1), Pb);
#pragma unroll
      for (int r = 0; r < 8; ++r)
        band[wave][r + 8 * hi][rt * 16 + (lane & 15)] = Pb[r] * scale;
    }
    asm volatile("s_wait_dscnt 0" ::: "memory");   // same-wave LDS WAR/RAW

    // ---- gather rel-shifted band + mask + scale ----
#pragma unroll
    for (int t = 0; t < 4; ++t) {
      const int jloc = t * 16 + (lane & 15);
      const int jg   = k0g + jloc;
#pragma unroll
      for (int r = 0; r < 8; ++r) {
        const int m  = r + 8 * hi;
        const int ig = q0w + m;
        const float rel = band[wave][m][jloc - m + 15];
        const unsigned char mk = maskp[((size_t)b * LSEQ + ig) * LSEQ + jg];
        const float s = S[t][r] * scale + rel;
        S[t][r] = mk ? -1e30f : s;
      }
    }

    // ---- online softmax update ----
#pragma unroll
    for (int r = 0; r < 8; ++r) {
      float mx = fmaxf(fmaxf(S[0][r], S[1][r]), fmaxf(S[2][r], S[3][r]));
      mx = redmax16(mx);
      const float mnew = fmaxf(mrun[r], mx);
      const float sc   = __expf(mrun[r] - mnew);
      mrun[r] = mnew;
      float rs = 0.f;
#pragma unroll
      for (int t = 0; t < 4; ++t) {
        const float p = __expf(S[t][r] - mnew);
        S[t][r] = p;
        rs += p;
      }
      rs = redsum16(rs);
      lrun[r] = lrun[r] * sc + rs;
#pragma unroll
      for (int t = 0; t < 4; ++t) O[t][r] *= sc;
    }

    // ---- restage P as bf16 (A-operand layout source) ----
#pragma unroll
    for (int t = 0; t < 4; ++t)
#pragma unroll
      for (int r = 0; r < 8; ++r)
        Pst[wave][(r + 8 * hi) * 64 + t * 16 + (lane & 15)] = (__bf16)S[t][r];
    asm volatile("s_wait_dscnt 0" ::: "memory");

    // ---- O += P * V ----
#pragma unroll
    for (int ks = 0; ks < 2; ++ks) {
      const v16bf ap = fragA_bf16(Pst[wave], 0, 64, 32 * ks);
#pragma unroll
      for (int t = 0; t < 4; ++t)
        O[t] = wmma_bf16(ap, fragB_kmajor_bf16(Vt, 32 * ks, HDIM, t * 16), O[t]);
    }
  }

  // ---- epilogue: normalize, write ctx as bf16 [B, L, E] ----
#pragma unroll
  for (int r = 0; r < 8; ++r) {
    const int m  = r + 8 * hi;
    const int ig = q0w + m;
    const float inv = (lrun[r] > 0.f) ? (1.f / lrun[r]) : 0.f;
#pragma unroll
    for (int t = 0; t < 4; ++t) {
      const int d = t * 16 + (lane & 15);
      ctx[((size_t)b * LSEQ + ig) * EDIM + h * HDIM + d] = (__bf16)(O[t][r] * inv);
    }
  }
}

// ---------------------------------------------------------------------------
extern "C" void kernel_launch(void* const* d_in, const int* in_sizes, int n_in,
                              void* d_out, int out_size, void* d_ws, size_t ws_size,
                              hipStream_t stream) {
  const float* query = (const float*)d_in[0];
  const float* key   = (const float*)d_in[1];
  const float* value = (const float*)d_in[2];
  const unsigned char* mask = (const unsigned char*)d_in[3];
  const float* Wq = (const float*)d_in[4];
  const float* bq = (const float*)d_in[5];
  const float* Wk = (const float*)d_in[6];
  const float* bk = (const float*)d_in[7];
  const float* Wv = (const float*)d_in[8];
  const float* bv = (const float*)d_in[9];
  const float* Wo = (const float*)d_in[10];
  const float* bo = (const float*)d_in[11];
  const float* rel = (const float*)d_in[12];
  const float* Wp  = (const float*)d_in[13];
  float* out = (float*)d_out;

  // workspace layout (bf16 intermediates, ~32.5 MB)
  char* ws = (char*)d_ws;
  __bf16* Qbf = (__bf16*)(ws);                                    // [B,H,L,D] 8 MB
  __bf16* Kbf = (__bf16*)(ws + (size_t)(8u << 20));               // 8 MB
  __bf16* Vbf = (__bf16*)(ws + (size_t)(16u << 20));              // 8 MB
  __bf16* Rk  = (__bf16*)(ws + (size_t)(24u << 20));              // 256 KB (+pad)
  __bf16* ctx = (__bf16*)(ws + (size_t)(24u << 20) + (1u << 19)); // [N,E] 8 MB

  const dim3 gGemm(NTOK / 128, EDIM / 64);   // (32, 16)

  relproj_kernel<<<(NREL + 15) / 16, 32, 0, stream>>>(rel, Wp, Rk);
  gemm_f32in_bf16out<<<gGemm, 256, 0, stream>>>(query, Wq, bq, Qbf);
  gemm_f32in_bf16out<<<gGemm, 256, 0, stream>>>(key,   Wk, bk, Kbf);
  gemm_f32in_bf16out<<<gGemm, 256, 0, stream>>>(value, Wv, bv, Vbf);
  attn_kernel<<<dim3(16, 64), 128, 0, stream>>>(Qbf, Kbf, Vbf, Rk, mask, ctx);
  gemm_bf16in_f32out<<<gGemm, 256, 0, stream>>>(ctx, Wo, bo, out);
}
